// MixHop_24481313587860
// MI455X (gfx1250) — matrix operand order repaired and checked by reading, not compiled, not run on hardware
//
#include <hip/hip_runtime.h>

typedef __attribute__((ext_vector_type(16))) _Float16 v16h;
typedef __attribute__((ext_vector_type(8)))  _Float16 v8h;
typedef __attribute__((ext_vector_type(8)))  float    v8f;

#define N_NODES 50000
#define E_EDGES 800000
#define IN_DIM  128
#define HIDDEN  256
#define OUT_DIM 40
#define N_TILES (N_NODES / 16)   // 3125 exactly

// Workspace layout (float element offsets)
#define AX_OFF   50048
#define AAX_OFF  (AX_OFF + N_NODES * IN_DIM)          // 6,450,048
#define PKF_OFF  (AAX_OFF + N_NODES * IN_DIM)         // 12,850,048 (floats)
// Packed weight fragment blocks (half element offsets within pk region)
#define PK_BLK   12288                                 // per-matrix frag halves
#define PK_W0    0
#define PK_W1    (1 * PK_BLK)
#define PK_W2    (2 * PK_BLK)
#define PK_WF    (3 * PK_BLK)
#define ZERO_V4  ((PKF_OFF) / 4)                       // 3,212,512 float4s

// ---------------------------------------------------------------- zero fill
__global__ void zero_ws_k(float* ws) {
    int i = blockIdx.x * blockDim.x + threadIdx.x;
    if (i < ZERO_V4) {
        float4 z = {0.f, 0.f, 0.f, 0.f};
        ((float4*)ws)[i] = z;
    }
}

// ------------------------------------------------------- pack B fragments
// Frag layout per matrix: idx = ((t*KC + kc)*32 + lane)*16 + j
//   n = t*16 + (lane&15) ; k = kc*32 + (j>>3)*16 + (lane>>4)*8 + (j&7)   (ISA 7.12.2 B layout)
__global__ void pack_weights_k(const float* W0, const float* W1,
                               const float* W2, const float* Wf,
                               _Float16* pk) {
    int tid = blockIdx.x * blockDim.x + threadIdx.x;
    if (tid >= 4 * PK_BLK) return;
    int which = tid / PK_BLK;
    int idx   = tid % PK_BLK;
    int j    = idx & 15;
    int lane = (idx >> 4) & 31;
    int rest = idx >> 9;               // t*KC + kc
    const float* W; int Ncols, KC;
    if (which == 0)      { W = W0; Ncols = 85; KC = 4; }
    else if (which == 1) { W = W1; Ncols = 85; KC = 4; }
    else if (which == 2) { W = W2; Ncols = 86; KC = 4; }
    else                 { W = Wf; Ncols = 40; KC = 8; }
    int kc = rest % KC;
    int t  = rest / KC;
    int n  = t * 16 + (lane & 15);
    int k  = kc * 32 + (j >> 3) * 16 + (lane >> 4) * 8 + (j & 7);
    float v = (n < Ncols) ? W[k * Ncols + n] : 0.0f;
    pk[tid] = (_Float16)v;
}

// ---------------------------------------------------------------- degree
__global__ void degree_k(const int* rows, float* deg) {
    int e = blockIdx.x * blockDim.x + threadIdx.x;
    if (e < E_EDGES) atomicAdd(&deg[rows[e]], 1.0f);
}

__global__ void deginv_k(float* deg) {
    int i = blockIdx.x * blockDim.x + threadIdx.x;
    if (i < N_NODES) {
        float d = deg[i];
        deg[i] = 1.0f / fmaxf(d, 1.0f);
    }
}

// ------------------------------------------------------------------ SpMM
// one wave per edge; lane l moves features [4l, 4l+4): b128 gather + 4 f32
// atomics that resolve in the 192MB L2 (dst is 25.6MB, fully resident).
__global__ void spmm_k(const int* rows, const int* cols,
                       const float* __restrict__ src, float* dst) {
    int tid = blockIdx.x * blockDim.x + threadIdx.x;
    int e = tid >> 5;
    int l = tid & 31;
    if (e >= E_EDGES) return;
    int c = cols[e];
    int r = rows[e];
    const float4 v = *(const float4*)(src + (long)c * IN_DIM + l * 4);
    float* d = dst + (long)r * IN_DIM + l * 4;
    atomicAdd(d + 0, v.x);
    atomicAdd(d + 1, v.y);
    atomicAdd(d + 2, v.z);
    atomicAdd(d + 3, v.w);
}

// --------------------------------------------------- fused dense (WMMA)
#define HSTRIDE 264            // 256 + pad, halves
#define WPB 2                  // waves per block

__global__ __launch_bounds__(32 * WPB)
void mixhop_dense_k(const float* __restrict__ x,  const float* __restrict__ ax,
                    const float* __restrict__ aax, const float* __restrict__ dinv,
                    const _Float16* __restrict__ pk,
                    const float* __restrict__ b0, const float* __restrict__ b1,
                    const float* __restrict__ b2, const float* __restrict__ bfin,
                    float* __restrict__ out) {
    __shared__ _Float16 hbuf[WPB][16][HSTRIDE];

    const int wave = threadIdx.x >> 5;
    const int lane = threadIdx.x & 31;
    const int tile = blockIdx.x * WPB + wave;
    if (tile >= N_TILES) return;            // whole wave uniform; no barriers used
    const int tb    = tile * 16;
    const int m     = lane & 15;            // row within tile (A frag / D col)
    const int half  = lane >> 4;            // lane half
    const int abase = half * 8;             // A-frag K interleave base

    const float dv = dinv[tb + m];
    _Float16 (*hrow)[HSTRIDE] = hbuf[wave];

    // ---------------- stage 1: h = relu([x@W0 | Ax*di@W1 | AAx*di^2@W2] + b)
    #pragma unroll
    for (int s = 0; s < 3; ++s) {
        const float* Asrc = (s == 0) ? x : (s == 1) ? ax : aax;
        const float  sc   = (s == 0) ? 1.0f : (s == 1) ? dv : dv * dv;
        const float* bias = (s == 0) ? b0 : (s == 1) ? b1 : b2;
        const int    hp   = (s == 2) ? 86 : 85;
        const int    off  = s * 85;
        const _Float16* pw = pk + s * PK_BLK;
        const float* Ap = Asrc + (long)(tb + m) * IN_DIM;

        v8f acc[6];
        #pragma unroll
        for (int t = 0; t < 6; ++t) { v8f z = {}; acc[t] = z; }

        #pragma unroll
        for (int kc = 0; kc < 4; ++kc) {
            // A fragment: rows striped per 16-bit A layout (ISA 7.12.2)
            float4 a0 = *(const float4*)(Ap + kc * 32 + abase);
            float4 a1 = *(const float4*)(Ap + kc * 32 + abase + 4);
            float4 a2 = *(const float4*)(Ap + kc * 32 + 16 + abase);
            float4 a3 = *(const float4*)(Ap + kc * 32 + 16 + abase + 4);
            v16h af;
            af[0]  = (_Float16)(a0.x * sc); af[1]  = (_Float16)(a0.y * sc);
            af[2]  = (_Float16)(a0.z * sc); af[3]  = (_Float16)(a0.w * sc);
            af[4]  = (_Float16)(a1.x * sc); af[5]  = (_Float16)(a1.y * sc);
            af[6]  = (_Float16)(a1.z * sc); af[7]  = (_Float16)(a1.w * sc);
            af[8]  = (_Float16)(a2.x * sc); af[9]  = (_Float16)(a2.y * sc);
            af[10] = (_Float16)(a2.z * sc); af[11] = (_Float16)(a2.w * sc);
            af[12] = (_Float16)(a3.x * sc); af[13] = (_Float16)(a3.y * sc);
            af[14] = (_Float16)(a3.z * sc); af[15] = (_Float16)(a3.w * sc);
            #pragma unroll
            for (int t = 0; t < 6; ++t) {
                v16h bfrag = *(const v16h*)(pw + (((t * 4 + kc) * 32) + lane) * 16);
                acc[t] = __builtin_amdgcn_wmma_f32_16x16x32_f16(
                    false, af, false, bfrag, (short)0, acc[t], false, false);
            }
        }
        // epilogue: bias + relu -> LDS (f16)
        #pragma unroll
        for (int t = 0; t < 6; ++t) {
            int col = t * 16 + m;
            if (col < hp) {
                float bv = bias[col];
                #pragma unroll
                for (int i = 0; i < 8; ++i) {
                    float v = acc[t][i] + bv;
                    v = v > 0.0f ? v : 0.0f;
                    hrow[half * 8 + i][off + col] = (_Float16)v;
                }
            }
        }
    }

    // ---------------- stage 2: out = h @ Wf + bf   (K=256 from LDS)
    v8f acc2[3];
    #pragma unroll
    for (int t = 0; t < 3; ++t) { v8f z = {}; acc2[t] = z; }

    const _Float16* pwf = pk + PK_WF;
    #pragma unroll
    for (int kc = 0; kc < 8; ++kc) {
        v8h lo = *(const v8h*)&hrow[m][kc * 32 + abase];
        v8h hi = *(const v8h*)&hrow[m][kc * 32 + 16 + abase];
        v16h af;
        #pragma unroll
        for (int j = 0; j < 8; ++j) { af[j] = lo[j]; af[8 + j] = hi[j]; }
        #pragma unroll
        for (int t = 0; t < 3; ++t) {
            v16h bfrag = *(const v16h*)(pwf + (((t * 8 + kc) * 32) + lane) * 16);
            acc2[t] = __builtin_amdgcn_wmma_f32_16x16x32_f16(
                false, af, false, bfrag, (short)0, acc2[t], false, false);
        }
    }
    #pragma unroll
    for (int t = 0; t < 3; ++t) {
        int col = t * 16 + m;
        if (col < OUT_DIM) {
            float bv = bfin[col];
            #pragma unroll
            for (int i = 0; i < 8; ++i)
                out[(long)(tb + half * 8 + i) * OUT_DIM + col] = acc2[t][i] + bv;
        }
    }
}

// ------------------------------------------------------------------ launch
extern "C" void kernel_launch(void* const* d_in, const int* in_sizes, int n_in,
                              void* d_out, int out_size, void* d_ws, size_t ws_size,
                              hipStream_t stream) {
    const float* x   = (const float*)d_in[0];
    const int*   ei  = (const int*)  d_in[1];
    const float* W0  = (const float*)d_in[2];
    const float* b0  = (const float*)d_in[3];
    const float* W1  = (const float*)d_in[4];
    const float* b1  = (const float*)d_in[5];
    const float* W2  = (const float*)d_in[6];
    const float* b2  = (const float*)d_in[7];
    const float* Wf  = (const float*)d_in[8];
    const float* bfv = (const float*)d_in[9];
    float* out = (float*)d_out;

    float*    ws  = (float*)d_ws;
    float*    deg = ws;                         // N floats (becomes deg_inv)
    float*    ax  = ws + AX_OFF;                // N x 128
    float*    aax = ws + AAX_OFF;               // N x 128
    _Float16* pk  = (_Float16*)(ws + PKF_OFF);  // packed B fragments

    const int* rows = ei;
    const int* cols = ei + E_EDGES;

    zero_ws_k<<<(ZERO_V4 + 255) / 256, 256, 0, stream>>>(ws);
    pack_weights_k<<<(4 * PK_BLK + 255) / 256, 256, 0, stream>>>(W0, W1, W2, Wf, pk);
    degree_k<<<(E_EDGES + 255) / 256, 256, 0, stream>>>(rows, deg);
    spmm_k<<<(E_EDGES * 32) / 256, 256, 0, stream>>>(rows, cols, x, ax);
    spmm_k<<<(E_EDGES * 32) / 256, 256, 0, stream>>>(rows, cols, ax, aax);
    deginv_k<<<(N_NODES + 255) / 256, 256, 0, stream>>>(deg);
    mixhop_dense_k<<<(N_TILES + WPB - 1) / WPB, 32 * WPB, 0, stream>>>(
        x, ax, aax, deg, pk, b0, b1, b2, bfv, out);
}
